// ResearchAccurateDiffusionModel_88862873355019
// MI455X (gfx1250) — matrix operand order repaired and checked by reading, not compiled, not run on hardware
//
#include <hip/hip_runtime.h>

#define HD 256          // hidden dim
#define RD 32           // radial dim
#define EF (2*HD + RD)  // 544 edge-feature width, 17 * 32
#define TE 16           // edges / nodes per block tile

typedef __attribute__((ext_vector_type(16))) __bf16 bf16x16;
typedef __attribute__((ext_vector_type(8)))  float  f32x8;

union FragU { unsigned u[8]; bf16x16 v; };

__device__ __forceinline__ unsigned short f2bf(float f) {
  unsigned u = __builtin_bit_cast(unsigned, f);
  u += 0x7FFFu + ((u >> 16) & 1u);          // round-to-nearest-even
  return (unsigned short)(u >> 16);
}
__device__ __forceinline__ float silu_f(float x) { return x / (1.0f + __expf(-x)); }

__device__ __forceinline__ f32x8 wmma_bf16(bf16x16 a, bf16x16 b, f32x8 c) {
  // D = A(16x32 bf16) * B(32x16 bf16) + C(16x16 f32)
  return __builtin_amdgcn_wmma_f32_16x16x32_bf16(false, a, false, b, (short)0, c, false, false);
}

// A-fragment from LDS tile [16][RS] bf16, rows = M, hardware 16-bit A layout:
// lane&15 = M row; lanes>=16 get K-base +8; VGPR j holds K pair {2j,2j+1} (+8 for j>=4)
template<int RS>
__device__ __forceinline__ bf16x16 load_a_frag(const unsigned short* lds, int lane, int k0) {
  FragU f;
  const unsigned short* row = lds + (lane & 15) * RS;
  int kb = k0 + ((lane >= 16) ? 8 : 0);
#pragma unroll
  for (int j = 0; j < 8; ++j) {
    int k = kb + 2 * j + ((j >= 4) ? 8 : 0);
    f.u[j] = *(const unsigned*)(row + k);   // RS even, k even -> 4B aligned
  }
  return f.v;
}

// B-fragment from pre-swizzled weights: frag (nt*nKt+kt) is 512 contiguous bf16,
// lane reads 16 contiguous elements (K run at fixed N)
__device__ __forceinline__ bf16x16 load_b_frag(const unsigned short* wp, int nKt,
                                               int kt, int nt, int lane) {
  FragU f;
  const unsigned short* p = wp + (((size_t)(nt * nKt + kt)) << 9) + lane * 16;
#pragma unroll
  for (int j = 0; j < 8; ++j) f.u[j] = *(const unsigned*)(p + 2 * j);
  return f.v;
}

// ---------------- prep kernels ----------------

__global__ void egnn_zero_kernel(float* __restrict__ a, int na,
                                 float* __restrict__ b, int nb) {
  int i = blockIdx.x * 256 + threadIdx.x;
  if (i < na) a[i] = 0.0f;
  if (i < nb) b[i] = 0.0f;
}

__global__ void egnn_h2bf_kernel(const float* __restrict__ h,
                                 unsigned short* __restrict__ hb, int n) {
  int i = blockIdx.x * 256 + threadIdx.x;
  if (i < n) hb[i] = f2bf(h[i]);
}

// Swizzle row-major f32 W[K][Nc] into bf16 WMMA B fragments.
// Fragment element e: lane=e/16, s=e%16 -> K = kt*32 + 8*(lane>=16) + s + 8*(s>=8), N = nt*16 + (lane&15)
__global__ void egnn_packw_kernel(const float* __restrict__ W,
                                  unsigned short* __restrict__ dst, int K, int Nc) {
  int i = blockIdx.x * 256 + threadIdx.x;
  int nKt = K / 32;
  int total = (K / 32) * (Nc / 16) * 512;
  if (i >= total) return;
  int frag = i >> 9, e = i & 511;
  int nt = frag / nKt, kt = frag % nKt;
  int lane = e >> 4, s = e & 15;
  int k = kt * 32 + ((lane >= 16) ? 8 : 0) + s + ((s >= 8) ? 8 : 0);
  int n = nt * 16 + (lane & 15);
  dst[i] = f2bf(W[(size_t)k * Nc + n]);
}

// ---------------- edge kernel: radial + edge MLP + LN + scatter + coord net ----------------

__global__ __launch_bounds__(128) void egnn_edge_kernel(
    const float* __restrict__ pos, const int* __restrict__ eidx, int E,
    const unsigned short* __restrict__ h_bf,
    const float* __restrict__ rb_w1, const float* __restrict__ rb_b1,
    const float* __restrict__ rb_w2, const float* __restrict__ rb_b2,
    const unsigned short* __restrict__ en_w1p, const float* __restrict__ en_b1,
    const unsigned short* __restrict__ en_w2p, const float* __restrict__ en_b2,
    const float* __restrict__ en_g, const float* __restrict__ en_bt,
    const unsigned short* __restrict__ cn_w1p, const float* __restrict__ cn_b1,
    const float* __restrict__ cn_w2, const float* __restrict__ cn_b2,
    float* __restrict__ node_msg, float* __restrict__ pos_out)
{
  __shared__ unsigned short feat[TE][EF];     // gathered edge features, bf16
  __shared__ unsigned short act[TE][HD];      // hidden act / edge_msg, bf16
  __shared__ float s_red[2][TE];              // LN sums
  __shared__ float s_rel[TE][3];
  __shared__ int   s_src[TE], s_dst[TE];
  __shared__ float s_cw[TE][HD / 2];          // coord-net hidden

  const int tid  = threadIdx.x;
  const int lane = tid & 31;
  const int wave = tid >> 5;
  const int mhi  = (lane & 16) >> 1;          // 0 or 8 (C/D row offset for high lanes)
  const int nlo  = lane & 15;

  if (tid < 2 * TE) ((float*)s_red)[tid] = 0.0f;
  if (tid < TE) {
    int e = blockIdx.x * TE + tid;
    int r = eidx[e], c = eidx[E + e];
    s_src[tid] = r; s_dst[tid] = c;
    float dx = pos[c * 3 + 0] - pos[r * 3 + 0];
    float dy = pos[c * 3 + 1] - pos[r * 3 + 1];
    float dz = pos[c * 3 + 2] - pos[r * 3 + 2];
    s_rel[tid][0] = dx; s_rel[tid][1] = dy; s_rel[tid][2] = dz;
    float ex = dx + 1e-8f, ey = dy + 1e-8f, ez = dz + 1e-8f;
    float d = sqrtf(ex * ex + ey * ey + ez * ez);
    float t1[RD];
#pragma unroll
    for (int i = 0; i < RD; ++i) t1[i] = silu_f(d * rb_w1[i] + rb_b1[i]);
    for (int o = 0; o < RD; ++o) {
      float a = rb_b2[o];
      for (int i = 0; i < RD; ++i) a += t1[i] * rb_w2[i * RD + o];
      feat[tid][2 * HD + o] = f2bf(a);
    }
  }
  __syncthreads();

  // gather h[row] || h[col] (bf16) into LDS tile
  for (int i = tid; i < TE * 2 * HD; i += 128) {
    int m = i >> 9, k = i & 511;
    int node = (k < HD) ? s_src[m] : s_dst[m];
    feat[m][k] = h_bf[(size_t)node * HD + (k & (HD - 1))];
  }
  __syncthreads();

  const f32x8 vzero = {0.f, 0.f, 0.f, 0.f, 0.f, 0.f, 0.f, 0.f};
  const unsigned short* featp = &feat[0][0];
  const unsigned short* actp  = &act[0][0];

  // ---- GEMM1: [16,544] x [544,256]; each wave owns 64 output cols ----
  f32x8 acc[4];
#pragma unroll
  for (int t = 0; t < 4; ++t) acc[t] = vzero;
  for (int kt = 0; kt < EF / 32; ++kt) {
    bf16x16 a = load_a_frag<EF>(featp, lane, kt * 32);
#pragma unroll
    for (int t = 0; t < 4; ++t)
      acc[t] = wmma_bf16(a, load_b_frag(en_w1p, EF / 32, kt, wave * 4 + t, lane), acc[t]);
  }
#pragma unroll
  for (int t = 0; t < 4; ++t) {
    int n = wave * 64 + t * 16 + nlo;
    float b = en_b1[n];
#pragma unroll
    for (int v = 0; v < 8; ++v) act[v + mhi][n] = f2bf(silu_f(acc[t][v] + b));
  }
  __syncthreads();

  // ---- GEMM2: [16,256] x [256,256] + bias ----
#pragma unroll
  for (int t = 0; t < 4; ++t) acc[t] = vzero;
  for (int kt = 0; kt < HD / 32; ++kt) {
    bf16x16 a = load_a_frag<HD>(actp, lane, kt * 32);
#pragma unroll
    for (int t = 0; t < 4; ++t)
      acc[t] = wmma_bf16(a, load_b_frag(en_w2p, HD / 32, kt, wave * 4 + t, lane), acc[t]);
  }
#pragma unroll
  for (int t = 0; t < 4; ++t) {
    float b = en_b2[wave * 64 + t * 16 + nlo];
#pragma unroll
    for (int v = 0; v < 8; ++v) acc[t][v] += b;
  }
  __syncthreads();

  // ---- LayerNorm over 256 cols (wave-distributed) ----
#pragma unroll
  for (int v = 0; v < 8; ++v) {
    float p = acc[0][v] + acc[1][v] + acc[2][v] + acc[3][v];
    float q = acc[0][v] * acc[0][v] + acc[1][v] * acc[1][v] +
              acc[2][v] * acc[2][v] + acc[3][v] * acc[3][v];
    atomicAdd(&s_red[0][v + mhi], p);
    atomicAdd(&s_red[1][v + mhi], q);
  }
  __syncthreads();
#pragma unroll
  for (int t = 0; t < 4; ++t) {
    int n = wave * 64 + t * 16 + nlo;
    float g = en_g[n], bt = en_bt[n];
#pragma unroll
    for (int v = 0; v < 8; ++v) {
      int m = v + mhi;
      float mu  = s_red[0][m] * (1.0f / HD);
      float var = s_red[1][m] * (1.0f / HD) - mu * mu;
      float val = (acc[t][v] - mu) * rsqrtf(var + 1e-5f) * g + bt;
      atomicAdd(&node_msg[(size_t)s_dst[m] * HD + n], val);  // segment_sum at col
      act[m][n] = f2bf(val);                                 // edge_msg for coord net
    }
  }
  __syncthreads();

  // ---- coord net GEMM: [16,256] x [256,128]; each wave owns 32 cols ----
  f32x8 cacc[2]; cacc[0] = vzero; cacc[1] = vzero;
  for (int kt = 0; kt < HD / 32; ++kt) {
    bf16x16 a = load_a_frag<HD>(actp, lane, kt * 32);
#pragma unroll
    for (int t = 0; t < 2; ++t)
      cacc[t] = wmma_bf16(a, load_b_frag(cn_w1p, HD / 32, kt, wave * 2 + t, lane), cacc[t]);
  }
#pragma unroll
  for (int t = 0; t < 2; ++t) {
    int n = wave * 32 + t * 16 + nlo;
    float b = cn_b1[n];
#pragma unroll
    for (int v = 0; v < 8; ++v) s_cw[v + mhi][n] = silu_f(cacc[t][v] + b);
  }
  __syncthreads();

  if (tid < TE) {
    float a = cn_b2[0];
    for (int i = 0; i < HD / 2; ++i) a += s_cw[tid][i] * cn_w2[i];
    float w = tanhf(a);
    int r = s_src[tid], c = s_dst[tid];
#pragma unroll
    for (int d = 0; d < 3; ++d) {
      float wd = s_rel[tid][d] * w;
      atomicAdd(&pos_out[(size_t)r * 3 + d], -wd);
      atomicAdd(&pos_out[(size_t)c * 3 + d],  wd);
    }
  }
}

// ---------------- node kernel: concat(h, node_msg) -> MLP -> LN ----------------

__global__ __launch_bounds__(128) void egnn_node_kernel(
    const unsigned short* __restrict__ h_bf,
    const float* __restrict__ node_msg,
    const unsigned short* __restrict__ nn_w1p, const float* __restrict__ nn_b1,
    const unsigned short* __restrict__ nn_w2p, const float* __restrict__ nn_b2,
    const float* __restrict__ nn_g, const float* __restrict__ nn_bt,
    float* __restrict__ h_out)
{
  __shared__ unsigned short feat[TE][2 * HD];
  __shared__ unsigned short act[TE][HD];
  __shared__ float s_red[2][TE];

  const int tid  = threadIdx.x;
  const int lane = tid & 31;
  const int wave = tid >> 5;
  const int mhi  = (lane & 16) >> 1;
  const int nlo  = lane & 15;
  const int n0   = blockIdx.x * TE;

  if (tid < 2 * TE) ((float*)s_red)[tid] = 0.0f;
  for (int i = tid; i < TE * 2 * HD; i += 128) {
    int m = i >> 9, k = i & 511;
    int node = n0 + m;
    feat[m][k] = (k < HD) ? h_bf[(size_t)node * HD + k]
                          : f2bf(node_msg[(size_t)node * HD + (k - HD)]);
  }
  __syncthreads();

  const f32x8 vzero = {0.f, 0.f, 0.f, 0.f, 0.f, 0.f, 0.f, 0.f};
  const unsigned short* featp = &feat[0][0];
  const unsigned short* actp  = &act[0][0];

  f32x8 acc[4];
#pragma unroll
  for (int t = 0; t < 4; ++t) acc[t] = vzero;
  for (int kt = 0; kt < (2 * HD) / 32; ++kt) {
    bf16x16 a = load_a_frag<2 * HD>(featp, lane, kt * 32);
#pragma unroll
    for (int t = 0; t < 4; ++t)
      acc[t] = wmma_bf16(a, load_b_frag(nn_w1p, (2 * HD) / 32, kt, wave * 4 + t, lane), acc[t]);
  }
#pragma unroll
  for (int t = 0; t < 4; ++t) {
    int n = wave * 64 + t * 16 + nlo;
    float b = nn_b1[n];
#pragma unroll
    for (int v = 0; v < 8; ++v) act[v + mhi][n] = f2bf(silu_f(acc[t][v] + b));
  }
  __syncthreads();

#pragma unroll
  for (int t = 0; t < 4; ++t) acc[t] = vzero;
  for (int kt = 0; kt < HD / 32; ++kt) {
    bf16x16 a = load_a_frag<HD>(actp, lane, kt * 32);
#pragma unroll
    for (int t = 0; t < 4; ++t)
      acc[t] = wmma_bf16(a, load_b_frag(nn_w2p, HD / 32, kt, wave * 4 + t, lane), acc[t]);
  }
#pragma unroll
  for (int t = 0; t < 4; ++t) {
    float b = nn_b2[wave * 64 + t * 16 + nlo];
#pragma unroll
    for (int v = 0; v < 8; ++v) acc[t][v] += b;
  }
  __syncthreads();
#pragma unroll
  for (int v = 0; v < 8; ++v) {
    float p = acc[0][v] + acc[1][v] + acc[2][v] + acc[3][v];
    float q = acc[0][v] * acc[0][v] + acc[1][v] * acc[1][v] +
              acc[2][v] * acc[2][v] + acc[3][v] * acc[3][v];
    atomicAdd(&s_red[0][v + mhi], p);
    atomicAdd(&s_red[1][v + mhi], q);
  }
  __syncthreads();
#pragma unroll
  for (int t = 0; t < 4; ++t) {
    int n = wave * 64 + t * 16 + nlo;
    float g = nn_g[n], bt = nn_bt[n];
#pragma unroll
    for (int v = 0; v < 8; ++v) {
      int m = v + mhi;
      float mu  = s_red[0][m] * (1.0f / HD);
      float var = s_red[1][m] * (1.0f / HD) - mu * mu;
      h_out[(size_t)(n0 + m) * HD + n] =
          (acc[t][v] - mu) * rsqrtf(var + 1e-5f) * g + bt;
    }
  }
}

// ---------------- launch ----------------

extern "C" void kernel_launch(void* const* d_in, const int* in_sizes, int n_in,
                              void* d_out, int out_size, void* d_ws, size_t ws_size,
                              hipStream_t stream) {
  const float* h     = (const float*)d_in[0];
  const float* pos   = (const float*)d_in[1];
  const float* rb_w1 = (const float*)d_in[2];
  const float* rb_b1 = (const float*)d_in[3];
  const float* rb_w2 = (const float*)d_in[4];
  const float* rb_b2 = (const float*)d_in[5];
  const float* en_w1 = (const float*)d_in[6];
  const float* en_b1 = (const float*)d_in[7];
  const float* en_w2 = (const float*)d_in[8];
  const float* en_b2 = (const float*)d_in[9];
  const float* en_g  = (const float*)d_in[10];
  const float* en_bt = (const float*)d_in[11];
  const float* nn_w1 = (const float*)d_in[12];
  const float* nn_b1 = (const float*)d_in[13];
  const float* nn_w2 = (const float*)d_in[14];
  const float* nn_b2 = (const float*)d_in[15];
  const float* nn_g  = (const float*)d_in[16];
  const float* nn_bt = (const float*)d_in[17];
  const float* cn_w1 = (const float*)d_in[18];
  const float* cn_b1 = (const float*)d_in[19];
  const float* cn_w2 = (const float*)d_in[20];
  const float* cn_b2 = (const float*)d_in[21];
  const int*   eidx  = (const int*)d_in[22];

  const int N = in_sizes[1] / 3;
  const int E = in_sizes[22] / 2;

  float* h_out   = (float*)d_out;
  float* pos_out = (float*)d_out + (size_t)N * HD;

  // workspace carve (aligned 256B)
  char* ws = (char*)d_ws;
  size_t ofs = 0;
  auto carve = [&](size_t bytes) -> void* {
    void* p = ws + ofs;
    ofs = (ofs + bytes + 255) & ~(size_t)255;
    return p;
  };
  float*          node_msg = (float*)carve((size_t)N * HD * 4);
  unsigned short* h_bf     = (unsigned short*)carve((size_t)N * HD * 2);
  unsigned short* en_w1p   = (unsigned short*)carve((size_t)EF * HD * 2);
  unsigned short* en_w2p   = (unsigned short*)carve((size_t)HD * HD * 2);
  unsigned short* nn_w1p   = (unsigned short*)carve((size_t)2 * HD * HD * 2);
  unsigned short* nn_w2p   = (unsigned short*)carve((size_t)HD * HD * 2);
  unsigned short* cn_w1p   = (unsigned short*)carve((size_t)HD * (HD / 2) * 2);
  (void)ws_size; (void)n_in; (void)out_size; (void)h;

  auto cdiv = [](int a, int b) { return (a + b - 1) / b; };

  // zero accumulators (every call; graph-replay safe)
  egnn_zero_kernel<<<cdiv(N * HD, 256), 256, 0, stream>>>(node_msg, N * HD, pos_out, N * 3);
  egnn_h2bf_kernel<<<cdiv(N * HD, 256), 256, 0, stream>>>(h, h_bf, N * HD);

  egnn_packw_kernel<<<cdiv(EF * HD, 256), 256, 0, stream>>>(en_w1, en_w1p, EF, HD);
  egnn_packw_kernel<<<cdiv(HD * HD, 256), 256, 0, stream>>>(en_w2, en_w2p, HD, HD);
  egnn_packw_kernel<<<cdiv(2 * HD * HD, 256), 256, 0, stream>>>(nn_w1, nn_w1p, 2 * HD, HD);
  egnn_packw_kernel<<<cdiv(HD * HD, 256), 256, 0, stream>>>(nn_w2, nn_w2p, HD, HD);
  egnn_packw_kernel<<<cdiv(HD * (HD / 2), 256), 256, 0, stream>>>(cn_w1, cn_w1p, HD, HD / 2);

  egnn_edge_kernel<<<E / TE, 128, 0, stream>>>(
      pos, eidx, E, h_bf,
      rb_w1, rb_b1, rb_w2, rb_b2,
      en_w1p, en_b1, en_w2p, en_b2, en_g, en_bt,
      cn_w1p, cn_b1, cn_w2, cn_b2,
      node_msg, pos_out);

  egnn_node_kernel<<<N / TE, 128, 0, stream>>>(
      h_bf, node_msg,
      nn_w1p, nn_b1, nn_w2p, nn_b2, nn_g, nn_bt,
      h_out);
}
